// GraphMatchingNetwork_56573309223898
// MI455X (gfx1250) — matrix operand order repaired and checked by reading, not compiled, not run on hardware
//
#include <hip/hip_runtime.h>

#define HID    128
#define NNODES 10000
#define NEDGES 320000

typedef __attribute__((ext_vector_type(16))) __bf16 v16bf;
typedef __attribute__((ext_vector_type(8)))  float  v8f;
typedef __attribute__((ext_vector_type(4)))  float  v4f;

enum { MODE_PLAIN = 0, MODE_CAT2 = 1, MODE_GATHER3 = 2 };

// ---------------------------------------------------------------------------
// Generic [M,K] x [K,128] GEMM with bias (+optional ReLU), bf16 WMMA, f32 acc.
// Workgroup = 256 threads = 8 waves; wave w computes rows [blk*128+16w, +16)
// across all 128 output columns (8 accumulator tiles of 16x16).
// A-source modes:
//   PLAIN  : A row r = A0[r, 0..K)
//   CAT2   : A row r = [A0[r,0..128) | A1[r,0..128)]                (K = 256)
//   GATHER3: A row r = [A0[idx0[r]] | A0[idx1[r]] | A2[r]]          (K = 384)
// Wt is the bf16 TRANSPOSED weight: Wt[n][k], n in [0,128), k in [0,K).
// ---------------------------------------------------------------------------
template <int MODE, bool RELU>
__global__ __launch_bounds__(256) void gemm128_wmma(
    const float* __restrict__ A0, const float* __restrict__ A1,
    const float* __restrict__ A2,
    const int* __restrict__ idx0, const int* __restrict__ idx1,
    const __bf16* __restrict__ Wt, const float* __restrict__ bias,
    float* __restrict__ C, int M, int K)
{
  const int lane    = threadIdx.x & 31;
  const int wave    = threadIdx.x >> 5;
  const int rowbase = blockIdx.x * 128 + wave * 16;
  if (rowbase >= M) return;                 // M is a multiple of 16
  const int r     = rowbase + (lane & 15);  // A-frag row for this lane
  const int khalf = (lane >> 4) << 4;       // A-frag K sub-offset: 0 or 16
  const int kq    = (lane >> 4) << 4;       // B-frag K sub-offset: 0 or 16

  int g0 = 0, g1 = 0;
  if (MODE == MODE_GATHER3) { g0 = idx0[r]; g1 = idx1[r]; }

  v8f acc[8];
#pragma unroll
  for (int i = 0; i < 8; ++i) acc[i] = v8f{};

  for (int kb = 0; kb < K; kb += 32) {
    // ---- A fragment: contiguous K layout (lane<16: K 0..15) ----
    v16bf afrag;
#pragma unroll
    for (int half = 0; half < 2; ++half) {
      const int k = kb + khalf + half * 8;
      const float* src;
      if (MODE == MODE_PLAIN) {
        src = A0 + (size_t)r * K + k;
      } else if (MODE == MODE_CAT2) {
        src = ((k >> 7) ? A1 : A0) + (size_t)r * HID + (k & 127);
      } else {
        const int seg = k >> 7;
        const float* base = (seg == 2) ? (A2 + (size_t)r * HID)
                                       : (A0 + (size_t)(seg ? g1 : g0) * HID);
        src = base + (k & 127);
      }
      v4f x0 = *(const v4f*)(src);
      v4f x1 = *(const v4f*)(src + 4);
#pragma unroll
      for (int j = 0; j < 4; ++j) {
        afrag[half * 8 + j]     = (__bf16)x0[j];
        afrag[half * 8 + 4 + j] = (__bf16)x1[j];
      }
    }
    // ---- B fragments + WMMA (lane<16: K kb..kb+15 of column n) ----
#pragma unroll
    for (int nt = 0; nt < 8; ++nt) {
      const int n = (nt << 4) + (lane & 15);
      v16bf bfrag = *(const v16bf*)(Wt + (size_t)n * K + kb + kq);
      acc[nt] = __builtin_amdgcn_wmma_f32_16x16x32_bf16(
          false, afrag, false, bfrag, (short)0, acc[nt], false, false);
    }
  }
  // ---- epilogue: C/D layout -> element (m = mlo+rr, n) ----
  const int mlo = (lane >> 4) << 3;
#pragma unroll
  for (int nt = 0; nt < 8; ++nt) {
    const int n  = (nt << 4) + (lane & 15);
    const float bv = bias[n];
#pragma unroll
    for (int rr = 0; rr < 8; ++rr) {
      float v = acc[nt][rr] + bv;
      if (RELU) v = fmaxf(v, 0.f);
      C[(size_t)(rowbase + mlo + rr) * HID + n] = v;
    }
  }
}

// fp32 W[K,128] -> bf16 Wt[128,K]  (transpose + convert)
__global__ __launch_bounds__(256) void convert_wT_kernel(
    const float* __restrict__ W, __bf16* __restrict__ Wt, int K)
{
  int t = blockIdx.x * blockDim.x + threadIdx.x;
  if (t >= K * HID) return;
  int k = t >> 7, n = t & 127;
  Wt[(size_t)n * K + k] = (__bf16)W[(size_t)k * HID + n];
}

__global__ __launch_bounds__(256) void zero_kernel(float* __restrict__ p, long n)
{
  long t = (long)blockIdx.x * blockDim.x + threadIdx.x;
  if (t < n) p[t] = 0.f;
}

// agg[rows[e], d] += msgs[e, d]   (relaxed agent-scope fp32 atomics)
__global__ __launch_bounds__(256) void scatter_add_kernel(
    const float* __restrict__ msgs, const int* __restrict__ rows,
    float* __restrict__ agg, long total)
{
  long t = (long)blockIdx.x * blockDim.x + threadIdx.x;
  if (t >= total) return;
  int e = (int)(t >> 7), d = (int)(t & 127);
  __hip_atomic_fetch_add(&agg[(size_t)rows[e] * HID + d], msgs[t],
                         __ATOMIC_RELAXED, __HIP_MEMORY_SCOPE_AGENT);
}

// Per-(node, head) cross attention: h <- h + softmax(q k^T/sqrt(32)) v + hinit
__global__ __launch_bounds__(256) void cross_attn_kernel(
    const float* __restrict__ q, const float* __restrict__ k,
    const float* __restrict__ v, const float* __restrict__ hinit,
    float* __restrict__ h, int Nn)
{
  int t = blockIdx.x * blockDim.x + threadIdx.x;
  int n = t >> 2, hh = t & 3;
  if (n >= Nn) return;
  const float* qp = q + (size_t)n * HID + hh * 32;
  float s[4], mx = -1e30f;
#pragma unroll
  for (int g = 0; g < 4; ++g) {
    const float* kp = k + (size_t)n * HID + g * 32;
    float d = 0.f;
#pragma unroll
    for (int j = 0; j < 32; ++j) d += qp[j] * kp[j];
    s[g] = d * 0.17677669529663687f;   // 1/sqrt(32)
    mx = fmaxf(mx, s[g]);
  }
  float den = 0.f;
#pragma unroll
  for (int g = 0; g < 4; ++g) { s[g] = __expf(s[g] - mx); den += s[g]; }
  const float inv = 1.f / den;
  float*       op = h     + (size_t)n * HID + hh * 32;
  const float* ip = hinit + (size_t)n * HID + hh * 32;
  const float* vp = v     + (size_t)n * HID;
#pragma unroll
  for (int j = 0; j < 32; ++j) {
    float u = 0.f;
#pragma unroll
    for (int g = 0; g < 4; ++g) u += s[g] * vp[g * 32 + j];
    op[j] = op[j] + u * inv + ip[j];
  }
}

// r[c] = sum_i h[i, c]
__global__ __launch_bounds__(256) void colsum_kernel(
    const float* __restrict__ h, float* __restrict__ r, int M)
{
  __shared__ float sm[256];
  const int c = blockIdx.x;
  float s = 0.f;
  for (int i = threadIdx.x; i < M; i += 256) s += h[(size_t)i * HID + c];
  sm[threadIdx.x] = s;
  __syncthreads();
  for (int st = 128; st > 0; st >>= 1) {
    if (threadIdx.x < st) sm[threadIdx.x] += sm[threadIdx.x + st];
    __syncthreads();
  }
  if (threadIdx.x == 0) r[c] = sm[0];
}

// out[0] = (r1 . W . r2) * invN2 + b      (mean of rank-1-collapsed bilinear)
__global__ __launch_bounds__(128) void score_kernel(
    const float* __restrict__ r1, const float* __restrict__ r2,
    const float* __restrict__ W, const float* __restrict__ b,
    float* __restrict__ out, float invN2)
{
  __shared__ float sm[128];
  const int j = threadIdx.x;
  float t = 0.f;
  for (int k = 0; k < HID; ++k) t += r1[k] * W[(size_t)k * HID + j];
  sm[j] = t * r2[j];
  __syncthreads();
  for (int st = 64; st > 0; st >>= 1) {
    if (j < st) sm[j] += sm[j + st];
    __syncthreads();
  }
  if (j == 0) out[0] = sm[0] * invN2 + b[0];
}

// ---------------------------------------------------------------------------
extern "C" void kernel_launch(void* const* d_in, const int* in_sizes, int n_in,
                              void* d_out, int out_size, void* d_ws, size_t ws_size,
                              hipStream_t stream)
{
  (void)in_sizes; (void)n_in; (void)out_size; (void)ws_size;
  const int N = NNODES, E = NEDGES;

  const float* x1     = (const float*)d_in[0];
  const int*   ei1    = (const int*)  d_in[1];
  const float* eattr1 = (const float*)d_in[2];
  const float* x2     = (const float*)d_in[3];
  const int*   ei2    = (const int*)  d_in[4];
  const float* eattr2 = (const float*)d_in[5];

  // params flattened in JAX pytree (sorted-key) order
  const int CROSS = 6;                // 3 iters x (k1,k2,q1,q2,v1,v2) x (w,b)
  const int EE    = CROSS + 36;       // edge_enc: w0,b0,w1,b1
  const int LAY   = EE + 4;           // 3 layers x (edge w0,b0,w1,b1, node w0,b0,w1,b1)
  const int NE    = LAY + 24;         // node_enc: w0,b0,w1,b1
  const int SIMW  = NE + 4;
  const int SIMB  = SIMW + 1;
  auto P = [&](int i) { return (const float*)d_in[i]; };

  // ---- workspace carve (256B aligned) ----
  char* base = (char*)d_ws;
  size_t off = 0;
  auto carve = [&](size_t bytes) -> void* {
    void* p = base + off; off += (bytes + 255) & ~(size_t)255; return p;
  };
  const size_t eb = (size_t)E * HID * sizeof(float);
  const size_t nb = (size_t)N * HID * sizeof(float);
  float* ea1  = (float*)carve(eb);
  float* ea2  = (float*)carve(eb);
  float* tmpE = (float*)carve(eb);
  float* msgs = (float*)carve(eb);
  float* hA1  = (float*)carve(nb);
  float* hB1  = (float*)carve(nb);
  float* hA2  = (float*)carve(nb);
  float* hB2  = (float*)carve(nb);
  float* h1i  = (float*)carve(nb);
  float* h2i  = (float*)carve(nb);
  float* agg  = (float*)carve(nb);
  float* tmpN = (float*)carve(nb);
  float* qb1  = (float*)carve(nb);
  float* kb1  = (float*)carve(nb);
  float* vb1  = (float*)carve(nb);
  float* qb2  = (float*)carve(nb);
  float* kb2  = (float*)carve(nb);
  float* vb2  = (float*)carve(nb);
  float* r1   = (float*)carve(512);
  float* r2   = (float*)carve(512);
  __bf16* Wt  = (__bf16*)carve((size_t)HID * 384 * sizeof(__bf16));

  auto cvt = [&](const float* w, int K) {
    convert_wT_kernel<<<(K * HID + 255) / 256, 256, 0, stream>>>(w, Wt, K);
  };
  auto linear = [&](const float* A, int wi, float* out, int M, bool relu) {
    cvt(P(wi), HID);
    dim3 g((M + 127) / 128);
    if (relu)
      gemm128_wmma<MODE_PLAIN, true ><<<g, 256, 0, stream>>>(
          A, nullptr, nullptr, nullptr, nullptr, Wt, P(wi + 1), out, M, HID);
    else
      gemm128_wmma<MODE_PLAIN, false><<<g, 256, 0, stream>>>(
          A, nullptr, nullptr, nullptr, nullptr, Wt, P(wi + 1), out, M, HID);
  };
  auto mlp2 = [&](const float* A, int base_i, float* scratch, float* out, int M) {
    linear(A, base_i, scratch, M, true);        // w0,b0 + ReLU
    linear(scratch, base_i + 2, out, M, false); // w1,b1
  };

  // ---- encoders ----
  mlp2(x1, NE, tmpN, hA1, N);
  mlp2(x2, NE, tmpN, hA2, N);
  mlp2(eattr1, EE, tmpE, ea1, E);
  mlp2(eattr2, EE, tmpE, ea2, E);

  // ---- GNN layers ----
  float* cur1 = hA1; float* alt1 = hB1;
  float* cur2 = hA2; float* alt2 = hB2;
  const long etotal = (long)E * HID;
  const int  egrid  = (int)((etotal + 255) / 256);
  for (int l = 0; l < 3; ++l) {
    const int lb = LAY + l * 8;
    for (int g = 0; g < 2; ++g) {
      float* h       = g ? cur2 : cur1;
      float* hn      = g ? alt2 : alt1;
      const int* row = g ? ei2 : ei1;        // edge_index[0]
      const int* col = (g ? ei2 : ei1) + E;  // edge_index[1]
      const float* ea = g ? ea2 : ea1;
      // edge MLP: relu([h[row]|h[col]|ea] @ W0 + b0) @ W1 + b1
      cvt(P(lb + 0), 3 * HID);
      gemm128_wmma<MODE_GATHER3, true><<<dim3(E / 128), 256, 0, stream>>>(
          h, nullptr, ea, row, col, Wt, P(lb + 1), tmpE, E, 3 * HID);
      linear(tmpE, lb + 2, msgs, E, false);
      // segment_sum over source node
      zero_kernel<<<(int)(((long)N * HID + 255) / 256), 256, 0, stream>>>(agg, (long)N * HID);
      scatter_add_kernel<<<egrid, 256, 0, stream>>>(msgs, row, agg, etotal);
      // node MLP: relu([h|agg] @ W0 + b0) @ W1 + b1
      cvt(P(lb + 4), 2 * HID);
      gemm128_wmma<MODE_CAT2, true><<<dim3((N + 127) / 128), 256, 0, stream>>>(
          h, agg, nullptr, nullptr, nullptr, Wt, P(lb + 5), tmpN, N, 2 * HID);
      linear(tmpN, lb + 6, hn, N, false);
    }
    { float* t = cur1; cur1 = alt1; alt1 = t; }
    { float* t = cur2; cur2 = alt2; alt2 = t; }
  }

  // ---- snapshot h1i/h2i, then cross-attention iterations (in place) ----
  hipMemcpyAsync(h1i, cur1, nb, hipMemcpyDeviceToDevice, stream);
  hipMemcpyAsync(h2i, cur2, nb, hipMemcpyDeviceToDevice, stream);
  const int agrid = (N * 4 + 255) / 256;
  for (int i = 0; i < 3; ++i) {
    const int cb = CROSS + i * 12;  // k1,k2,q1,q2,v1,v2 (w,b each)
    linear(cur1, cb + 0, kb1, N, false);  // k1(h1)
    linear(cur2, cb + 2, kb2, N, false);  // k2(h2)
    linear(cur1, cb + 4, qb1, N, false);  // q1(h1)
    linear(cur2, cb + 6, qb2, N, false);  // q2(h2)
    linear(cur1, cb + 8, vb1, N, false);  // v1(h1)
    linear(cur2, cb + 10, vb2, N, false); // v2(h2)
    cross_attn_kernel<<<agrid, 256, 0, stream>>>(qb1, kb2, vb2, h1i, cur1, N);
    cross_attn_kernel<<<agrid, 256, 0, stream>>>(qb2, kb1, vb1, h2i, cur2, N);
  }

  // ---- score = mean(h1 @ W @ h2^T) + b = (colsum(h1).W.colsum(h2))/N^2 + b ----
  colsum_kernel<<<HID, 256, 0, stream>>>(cur1, r1, N);
  colsum_kernel<<<HID, 256, 0, stream>>>(cur2, r2, N);
  float* out = (float*)d_out;
  score_kernel<<<1, 128, 0, stream>>>(r1, r2, P(SIMW), P(SIMB), out,
                                      1.0f / ((float)N * (float)N));
  hipMemcpyAsync(out + 1, cur1, nb, hipMemcpyDeviceToDevice, stream);
}
